// ImplicitNetwork_19370302505666
// MI455X (gfx1250) — matrix operand order repaired and checked by reading, not compiled, not run on hardware
//
#include <hip/hip_runtime.h>
#include <hip/hip_bf16.h>
#include <math.h>
#include <stdint.h>

typedef __attribute__((ext_vector_type(16))) _Float16 v16h;
typedef __attribute__((ext_vector_type(8)))  _Float16 v8h;
typedef __attribute__((ext_vector_type(8)))  float    v8f;

#define MT   64      // rows per block
#define AP   264     // activation LDS row stride (halves): 256 + 8 pad
#define KPW  264     // weight-chunk LDS row stride (halves)

struct MlpParams {
    const _Float16* W[9];
    const float*    B[9];
};

// ---------------------------------------------------------------------------
// Preprocess one layer: W16[i][k] = g[i]/||V[i,:]|| * extra * V[i][k]  (f16),
// zero-padded to [Opad][Kpad]; bias copied f32 (zero-padded).
// grid = Opad blocks, 256 threads.
// ---------------------------------------------------------------------------
__global__ void prep_wn(const float* __restrict__ V, const float* __restrict__ g,
                        const float* __restrict__ b,
                        _Float16* __restrict__ Wout, float* __restrict__ Bout,
                        int O, int K, int Kpad, float extra) {
    __shared__ float red[256];
    const int i = blockIdx.x;
    const int t = threadIdx.x;
    if (i >= O) {
        for (int k = t; k < Kpad; k += 256) Wout[(size_t)i * Kpad + k] = (_Float16)0.f;
        if (t == 0) Bout[i] = 0.f;
        return;
    }
    float p = 0.f;
    for (int k = t; k < K; k += 256) { float v = V[(size_t)i * K + k]; p += v * v; }
    red[t] = p;
    __syncthreads();
    for (int s = 128; s > 0; s >>= 1) {
        if (t < s) red[t] += red[t + s];
        __syncthreads();
    }
    const float scale = g[i] * rsqrtf(red[0]) * extra;
    for (int k = t; k < Kpad; k += 256)
        Wout[(size_t)i * Kpad + k] = (k < K) ? (_Float16)(V[(size_t)i * K + k] * scale)
                                             : (_Float16)0.f;
    if (t == 0) Bout[i] = b[i];
}

// ---------------------------------------------------------------------------
// Async staging of a 64-output-row weight chunk into LDS via the CDNA5
// GLOBAL_LOAD_ASYNC_TO_LDS path (tracked by ASYNCcnt, no VGPR round-trip).
// ---------------------------------------------------------------------------
__device__ __forceinline__ void stage_chunk_async(const _Float16* __restrict__ Wl,
                                                  int Kpad, int nt0,
                                                  _Float16* wdst, int tid) {
    const int vpr   = Kpad >> 3;      // v8h (16B) vectors per row
    const int total = 64 * vpr;       // multiple of 128
    for (int idx = tid; idx < total; idx += 128) {
        const int rloc = idx / vpr;
        const int ko   = (idx - rloc * vpr) << 3;
        // flat shared pointer keeps the LDS byte offset in addr[31:0]
        const unsigned lds  = (unsigned)(uintptr_t)&wdst[rloc * KPW + ko];
        const unsigned goff = (unsigned)(((nt0 * 16 + rloc) * Kpad + ko) * 2);
        asm volatile("global_load_async_to_lds_b128 %0, %1, %2"
                     :: "v"(lds), "v"(goff), "s"(Wl)
                     : "memory");
    }
}

#define ASYNC_WAIT() asm volatile("s_wait_asynccnt 0x0" ::: "memory")

// ---------------------------------------------------------------------------
// Fully fused MLP: 128 threads = 4 waves, 64 rows per block.
// ---------------------------------------------------------------------------
__global__ __launch_bounds__(128)
void ImplicitNetwork_19370302505666_kernel(const float* __restrict__ x, int N,
                                           MlpParams p, float* __restrict__ out) {
    __shared__ _Float16 act[2][MT * AP];    // ping-pong activations (f16)
    __shared__ _Float16 emb[MT * 40];       // positional embedding stash (39 used)
    __shared__ _Float16 wbuf[2][64 * KPW];  // double-buffered weight chunks

    const int tid  = threadIdx.x;
    const int lane = tid & 31;
    const int wave = tid >> 5;
    const long r0  = (long)blockIdx.x * MT;

    // ---- positional embedding: [x, sin(2^k x)(3), cos(2^k x)(3)] k=0..5 -> 39
    if (tid < MT) {
        const long row = r0 + tid;
        float x0 = 0.f, x1 = 0.f, x2 = 0.f;
        if (row < N) { x0 = x[row * 3 + 0]; x1 = x[row * 3 + 1]; x2 = x[row * 3 + 2]; }
        _Float16* e = &emb[tid * 40];
        e[0] = (_Float16)x0; e[1] = (_Float16)x1; e[2] = (_Float16)x2;
        float f = 1.f;
        #pragma unroll
        for (int k = 0; k < 6; ++k) {
            e[3 + k * 6 + 0] = (_Float16)__sinf(x0 * f);
            e[3 + k * 6 + 1] = (_Float16)__sinf(x1 * f);
            e[3 + k * 6 + 2] = (_Float16)__sinf(x2 * f);
            e[3 + k * 6 + 3] = (_Float16)__cosf(x0 * f);
            e[3 + k * 6 + 4] = (_Float16)__cosf(x1 * f);
            e[3 + k * 6 + 5] = (_Float16)__cosf(x2 * f);
            f *= 2.f;
        }
        _Float16* a = &act[0][tid * AP];
        #pragma unroll
        for (int j = 0; j < 39; ++j) a[j] = e[j];
        #pragma unroll
        for (int j = 39; j < 64; ++j) a[j] = (_Float16)0.f;
    }

    const int kitersA[9] = {2, 8, 8, 8, 8, 8, 8, 8, 8};
    const int ntilesA[9] = {16, 16, 16, 16, 16, 16, 16, 16, 20};
    const int KpadA[9]   = {64, 256, 256, 256, 256, 256, 256, 256, 256};

    int cur = 0;
    int buf = 0;
    for (int l = 0; l < 9; ++l) {
        const _Float16* Wl = p.W[l];
        const float*    Bl = p.B[l];
        const int kiters = kitersA[l];
        const int ntiles = ntilesA[l];
        const int Kpad   = KpadA[l];
        const _Float16* Ab = &act[cur][0];
        _Float16*       Ob = &act[cur ^ 1][0];
        const bool last = (l == 8);

        // per-lane fragment addressing (wave-uniform where needed)
        const int arow  = (wave * 16 + (lane & 15)) * AP;
        const int aoffL = (lane & 16) ? 8 : 0;     // A: hi half-wave holds K+8
        const int boffL = (lane & 16) ? 8 : 0;     // B: hi half-wave holds K+8
        const int brow  = (lane & 15) * KPW;
        const int mrow  = wave * 16 + ((lane & 16) ? 8 : 0);

        for (int nt0 = 0; nt0 < ntiles; nt0 += 4) {
            if (nt0 == 0) {
                // first chunk of the layer: stage synchronously
                stage_chunk_async(Wl, Kpad, 0, &wbuf[buf][0], tid);
                ASYNC_WAIT();
                __syncthreads();
            }
            const bool pf = (nt0 + 4 < ntiles);
            if (pf)  // overlap next chunk's DMA with this chunk's WMMAs
                stage_chunk_async(Wl, Kpad, nt0 + 4, &wbuf[buf ^ 1][0], tid);

            const _Float16* Wb = &wbuf[buf][0];

            // ---- 4 accumulator tiles (16x16 f32 each)
            v8f acc[4] = {};
            for (int kt = 0; kt < kiters; ++kt) {
                const int kb = kt * 32;
                const v8h alo = *(const v8h*)&Ab[arow + kb + aoffL];
                const v8h ahi = *(const v8h*)&Ab[arow + kb + aoffL + 16];
                const v16h af = __builtin_shufflevector(alo, ahi,
                    0, 1, 2, 3, 4, 5, 6, 7, 8, 9, 10, 11, 12, 13, 14, 15);
                #pragma unroll
                for (int j = 0; j < 4; ++j) {
                    const _Float16* bp = &Wb[(j * 16) * KPW + brow + kb + boffL];
                    const v8h blo = *(const v8h*)bp;
                    const v8h bhi = *(const v8h*)(bp + 16);
                    const v16h bf = __builtin_shufflevector(blo, bhi,
                        0, 1, 2, 3, 4, 5, 6, 7, 8, 9, 10, 11, 12, 13, 14, 15);
                    acc[j] = __builtin_amdgcn_wmma_f32_16x16x32_f16(
                        false, af, false, bf, (short)0, acc[j], false, false);
                }
            }

            // ---- bias + softplus(beta=100) + store
            #pragma unroll
            for (int j = 0; j < 4; ++j) {
                const int col = (nt0 + j) * 16 + (lane & 15);
                const float bias = Bl[col];
                #pragma unroll
                for (int r = 0; r < 8; ++r) {
                    const float v = acc[j][r] + bias;
                    if (!last) {
                        const float z  = 100.f * v;
                        const float zc = (z < 20.f) ? z : 20.f;
                        const float sp = (z > 20.f) ? v : log1pf(__expf(zc)) * 0.01f;
                        Ob[(mrow + r) * AP + col] = (_Float16)sp;
                    } else {
                        const long row = r0 + mrow + r;
                        if (col < 260 && row < N) out[row * 260 + col] = v;
                    }
                }
            }

            if (pf) ASYNC_WAIT();   // own prefetch landed before the barrier
            __syncthreads();        // cross-wave visibility (wbuf + activations)
            if (pf) buf ^= 1;
        }

        // ---- skip-concat before layer 4: overwrite cols 217..255 with emb
        // (the 1/sqrt(2) scale is folded into W4 during preprocessing)
        if (l == 3) {
            if (tid < MT) {
                _Float16*       a = &act[cur ^ 1][tid * AP];
                const _Float16* e = &emb[tid * 40];
                #pragma unroll
                for (int j = 0; j < 39; ++j) a[217 + j] = e[j];
            }
        }
        cur ^= 1;
    }
}

// ---------------------------------------------------------------------------
// Host launcher
// ---------------------------------------------------------------------------
extern "C" void kernel_launch(void* const* d_in, const int* in_sizes, int n_in,
                              void* d_out, int out_size, void* d_ws, size_t ws_size,
                              hipStream_t stream) {
    const float* x = (const float*)d_in[0];
    const int N = in_sizes[0] / 3;

    struct LDef { int O, K, Opad, Kpad; };
    const LDef L[9] = {
        {256,  39, 256,  64}, {256, 256, 256, 256}, {256, 256, 256, 256},
        {217, 256, 256, 256}, {256, 256, 256, 256}, {256, 256, 256, 256},
        {256, 256, 256, 256}, {256, 256, 256, 256}, {260, 256, 320, 256}};

    char* ws = (char*)d_ws;
    size_t off = 0;
    MlpParams p;
    _Float16* W[9];
    float*    Bb[9];
    for (int l = 0; l < 9; ++l) {
        W[l] = (_Float16*)(ws + off);
        off += (size_t)L[l].Opad * L[l].Kpad * sizeof(_Float16);
        off  = (off + 255) & ~(size_t)255;
    }
    for (int l = 0; l < 9; ++l) {
        Bb[l] = (float*)(ws + off);
        off += (size_t)L[l].Opad * sizeof(float);
        off  = (off + 255) & ~(size_t)255;
    }
    if (off > ws_size) return;  // insufficient scratch (not expected)

    for (int l = 0; l < 9; ++l) {
        const float extra = (l == 4) ? 0.70710678118654752f : 1.0f;
        prep_wn<<<L[l].Opad, 256, 0, stream>>>(
            (const float*)d_in[1 + 3 * l], (const float*)d_in[2 + 3 * l],
            (const float*)d_in[3 + 3 * l], W[l], Bb[l],
            L[l].O, L[l].K, L[l].Kpad, extra);
        p.W[l] = W[l];
        p.B[l] = Bb[l];
    }

    const int blocks = (N + MT - 1) / MT;
    ImplicitNetwork_19370302505666_kernel<<<blocks, 128, 0, stream>>>(
        x, N, p, (float*)d_out);
}
